// CausalAttention_12601434046486
// MI455X (gfx1250) — matrix operand order repaired and checked by reading, not compiled, not run on hardware
//
#include <hip/hip_runtime.h>
#include <hip/hip_bf16.h>
#include <hip/hip_fp16.h>

// Problem constants (match reference)
#define B_     2
#define S_     2048
#define HID_   2048
#define H_     16
#define KVH_   4
#define D_     128
#define GROUPS_ (H_ / KVH_)
#define MTOT_  (B_ * S_)        // 4096 token rows

typedef __attribute__((ext_vector_type(16))) _Float16 v16h;
typedef __attribute__((ext_vector_type(8)))  float    v8f;

static __device__ inline v8f v8f_zero() {
  v8f z = {0.f, 0.f, 0.f, 0.f, 0.f, 0.f, 0.f, 0.f};
  return z;
}

// ---------------------------------------------------------------------------
// WMMA fragment helpers, per CDNA5 ISA 7.12.2 (wave32):
//  A (16x32 f16): lane L holds row M=L%16; K-base kb=(L/16)*8; halves 0..7 are
//                 K=kb..kb+7, halves 8..15 are K=kb+16..kb+23.
//  B (32x16 f16): lane L holds col N=L%16; K-base kb=(L/16)*8; halves 0..7 are
//                 K=kb..kb+7, halves 8..15 are K=kb+16..kb+23.
//  C/D (16x16 f32): lane L holds col N=L%16, rows M=(L/16)*8 + i (vgpr i).
// All loaders below are per-lane CONTIGUOUS (b128-able) in the source.
// ---------------------------------------------------------------------------

static __device__ inline v16h load_a_f16(const _Float16* __restrict__ A, int lda, int lane) {
  const int m  = lane & 15;
  const int kb = (lane >> 4) << 3;
  const _Float16* p = A + (size_t)m * lda + kb;
  v16h a;
#pragma unroll
  for (int i = 0; i < 8; ++i) a[i] = p[i];
#pragma unroll
  for (int i = 0; i < 8; ++i) a[8 + i] = p[16 + i];
  return a;
}

// B fragment where logical B[k][n] = Src[n*lds + k] (Src is N-major / K-contig).
static __device__ inline v16h load_bT_f16(const _Float16* __restrict__ Sp, int lds_, int lane) {
  const int n  = lane & 15;
  const int kb = (lane >> 4) << 3;
  const _Float16* p = Sp + (size_t)n * lds_ + kb;
  v16h b;
#pragma unroll
  for (int e = 0; e < 8; ++e) { b[e] = p[e]; b[8 + e] = p[16 + e]; }
  return b;
}

static __device__ inline void store_c_f16(_Float16* __restrict__ C, int ldc, int lane, v8f c) {
  const int n  = lane & 15;
  const int mb = (lane >> 4) << 3;
  _Float16* p = C + (size_t)mb * ldc + n;
#pragma unroll
  for (int i = 0; i < 8; ++i) p[(size_t)i * ldc] = (_Float16)c[i];
}

// Transposed C store: writes Ct[n][m] (ldc = M); per lane one 16B run.
static __device__ inline void store_c_f16_T(_Float16* __restrict__ Ct, int ldc, int lane, v8f c) {
  const int n  = lane & 15;
  const int mb = (lane >> 4) << 3;
  _Float16* p = Ct + (size_t)n * ldc + mb;
#pragma unroll
  for (int i = 0; i < 8; ++i) p[i] = (_Float16)c[i];
}

static __device__ inline void store_c_f32(float* __restrict__ C, int ldc, int lane, v8f c) {
  const int n  = lane & 15;
  const int mb = (lane >> 4) << 3;
  float* p = C + (size_t)mb * ldc + n;
#pragma unroll
  for (int i = 0; i < 8; ++i) p[(size_t)i * ldc] = c[i];
}

// ---------------------------------------------------------------------------
// Pre-pass: f32 -> f16 elementwise conversion (coalesced, 4 elems/thread).
// ---------------------------------------------------------------------------
__global__ void cvt_f32_to_f16_kernel(const float* __restrict__ s, _Float16* __restrict__ d,
                                      size_t n) {
  const size_t i = ((size_t)blockIdx.x * blockDim.x + threadIdx.x) * 4;
  if (i + 3 < n) {
    const float4 v = *(const float4*)(s + i);
    d[i]     = (_Float16)v.x;
    d[i + 1] = (_Float16)v.y;
    d[i + 2] = (_Float16)v.z;
    d[i + 3] = (_Float16)v.w;
  }
}

// ---------------------------------------------------------------------------
// Pre-pass: transpose + convert weights. W is (K x N) f32 row-major,
// Wt is (N x K) f16 row-major. LDS-tiled so both sides are coalesced.
// ---------------------------------------------------------------------------
__global__ void __launch_bounds__(256)
transpose_cvt_kernel(const float* __restrict__ W, _Float16* __restrict__ Wt, int K, int N) {
  __shared__ _Float16 t[32][33];
  const int n0 = blockIdx.x * 32;
  const int k0 = blockIdx.y * 32;
  const int tx = threadIdx.x;
  const int ty = threadIdx.y;
#pragma unroll
  for (int r = 0; r < 32; r += 8)
    t[ty + r][tx] = (_Float16)W[(size_t)(k0 + ty + r) * N + n0 + tx];
  __syncthreads();
#pragma unroll
  for (int r = 0; r < 32; r += 8)
    Wt[(size_t)(n0 + ty + r) * K + k0 + tx] = t[tx][ty + r];
}

// ---------------------------------------------------------------------------
// GEMM: C[M,N] = A_f16[M,K] @ Bt_f16[N,K]^T   (f32 accumulation in WMMA)
// One wave per block; 64x64 tile per wave: 4 A-frags x 4 B-frags = 16 WMMAs
// per K-step of 32 on 16 contiguous b128 loads (each operand reused 4x).
// STORE_T=true writes C transposed (produces V^T for attention).
// ---------------------------------------------------------------------------
template <bool STORE_T>
__global__ void __launch_bounds__(32)
gemm_f16T_f16(const _Float16* __restrict__ A, const _Float16* __restrict__ Bt,
              _Float16* __restrict__ C, int M, int N, int K) {
  const int lane = threadIdx.x & 31;
  const int n0 = blockIdx.x * 64;
  const int m0 = blockIdx.y * 64;
  v8f acc[4][4];
#pragma unroll
  for (int r = 0; r < 4; ++r)
#pragma unroll
    for (int t = 0; t < 4; ++t) acc[r][t] = v8f_zero();

  for (int k0 = 0; k0 < K; k0 += 32) {
    if (k0 + 32 < K) {  // prefetch next K-step (global_prefetch_b8)
      __builtin_prefetch(A + (size_t)(m0 + (lane & 15)) * K + k0 + 32, 0, 1);
      __builtin_prefetch(Bt + (size_t)(n0 + (lane & 15)) * K + k0 + 32, 0, 1);
    }
    v16h a[4];
#pragma unroll
    for (int r = 0; r < 4; ++r)
      a[r] = load_a_f16(A + (size_t)(m0 + r * 16) * K + k0, K, lane);
#pragma unroll
    for (int t = 0; t < 4; ++t) {
      v16h b = load_bT_f16(Bt + (size_t)(n0 + t * 16) * K + k0, K, lane);
#pragma unroll
      for (int r = 0; r < 4; ++r)
        acc[r][t] = __builtin_amdgcn_wmma_f32_16x16x32_f16(false, a[r], false, b,
                                                           (short)0, acc[r][t], false, false);
    }
  }
#pragma unroll
  for (int r = 0; r < 4; ++r)
#pragma unroll
    for (int t = 0; t < 4; ++t) {
      if (STORE_T)
        store_c_f16_T(C + (size_t)(n0 + t * 16) * M + m0 + r * 16, M, lane, acc[r][t]);
      else
        store_c_f16(C + (size_t)(m0 + r * 16) * N + n0 + t * 16, N, lane, acc[r][t]);
    }
}

// Same tiling, f32 output (final output projection).
__global__ void __launch_bounds__(32)
gemm_f16T_f32(const _Float16* __restrict__ A, const _Float16* __restrict__ Bt,
              float* __restrict__ C, int M, int N, int K) {
  const int lane = threadIdx.x & 31;
  const int n0 = blockIdx.x * 64;
  const int m0 = blockIdx.y * 64;
  v8f acc[4][4];
#pragma unroll
  for (int r = 0; r < 4; ++r)
#pragma unroll
    for (int t = 0; t < 4; ++t) acc[r][t] = v8f_zero();

  for (int k0 = 0; k0 < K; k0 += 32) {
    if (k0 + 32 < K) {
      __builtin_prefetch(A + (size_t)(m0 + (lane & 15)) * K + k0 + 32, 0, 1);
      __builtin_prefetch(Bt + (size_t)(n0 + (lane & 15)) * K + k0 + 32, 0, 1);
    }
    v16h a[4];
#pragma unroll
    for (int r = 0; r < 4; ++r)
      a[r] = load_a_f16(A + (size_t)(m0 + r * 16) * K + k0, K, lane);
#pragma unroll
    for (int t = 0; t < 4; ++t) {
      v16h b = load_bT_f16(Bt + (size_t)(n0 + t * 16) * K + k0, K, lane);
#pragma unroll
      for (int r = 0; r < 4; ++r)
        acc[r][t] = __builtin_amdgcn_wmma_f32_16x16x32_f16(false, a[r], false, b,
                                                           (short)0, acc[r][t], false, false);
    }
  }
#pragma unroll
  for (int r = 0; r < 4; ++r)
#pragma unroll
    for (int t = 0; t < 4; ++t)
      store_c_f32(C + (size_t)(m0 + r * 16) * N + n0 + t * 16, N, lane, acc[r][t]);
}

// ---------------------------------------------------------------------------
// RoPE (in place on f16 [B*S, nheads*D]); pairs (2i, 2i+1) within each head.
// freqs_cis is (S, D/2, 2) f32: [s][i][0]=cos, [s][i][1]=sin.
// ---------------------------------------------------------------------------
__global__ void rope_f16_kernel(_Float16* __restrict__ t, const float* __restrict__ fc,
                                int nheads) {
  const size_t idx = (size_t)blockIdx.x * blockDim.x + threadIdx.x;
  const size_t npairs = (size_t)MTOT_ * nheads * (D_ / 2);
  if (idx >= npairs) return;
  const int i = (int)(idx % (D_ / 2));
  const size_t r = idx / (D_ / 2);
  const int h = (int)(r % nheads);
  const size_t bs = r / nheads;   // b*S + s
  const int s = (int)(bs % S_);
  const float c  = fc[(size_t)s * D_ + 2 * i];
  const float sn = fc[(size_t)s * D_ + 2 * i + 1];
  _Float16* p = t + bs * ((size_t)nheads * D_) + (size_t)h * D_ + 2 * i;
  const float xr = (float)p[0];
  const float xi = (float)p[1];
  p[0] = (_Float16)(xr * c - xi * sn);
  p[1] = (_Float16)(xr * sn + xi * c);
}

// ---------------------------------------------------------------------------
// Flash attention (causal, GQA). One wave per 16-row Q tile, 64 keys/step.
// Q: [B*S, H*D] f16 (post-RoPE), K: [B*S, KVH*D] f16 (post-RoPE),
// Vt: [KVH*D, B*S] f16 (produced transposed by the V projection GEMM),
// O: [B*S, H*D] f16.
// ---------------------------------------------------------------------------
__global__ void __launch_bounds__(32)
flash_attn_kernel(const _Float16* __restrict__ Q, const _Float16* __restrict__ Kt,
                  const _Float16* __restrict__ Vt, _Float16* __restrict__ O) {
  __shared__ _Float16 pl[16 * 64];   // P tile staging for A-fragment relayout
  const int lane = threadIdx.x & 31;
  const int qt = blockIdx.x;          // S/16 tiles
  const int h  = blockIdx.y;          // H
  const int b  = blockIdx.z;          // B
  const int kvh = h / GROUPS_;
  const int m0 = qt * 16;
  const int ldq = H_ * D_;
  const int ldk = KVH_ * D_;
  const _Float16* qb = Q  + ((size_t)b * S_ + m0) * ldq + h * D_;
  const _Float16* kb = Kt + (size_t)b * S_ * ldk + kvh * D_;
  // V^T base for this (b, kvh): Vt[(kvh*D + d)][b*S + s], row stride MTOT_
  const _Float16* vtb = Vt + (size_t)(kvh * D_) * MTOT_ + (size_t)b * S_;

  const int nidx  = lane & 15;          // C-frag column within tile
  const int rhalf = (lane >> 4) << 3;   // C-frag row base within tile

  // Preload all Q A-fragments for D=128 (4 K-chunks of 32)
  v16h qf[4];
#pragma unroll
  for (int kc = 0; kc < 4; ++kc) qf[kc] = load_a_f16(qb + kc * 32, ldq, lane);

  v8f acc[8];
#pragma unroll
  for (int dt = 0; dt < 8; ++dt) acc[dt] = v8f_zero();
  float mrow[8], lrow[8];
#pragma unroll
  for (int i = 0; i < 8; ++i) { mrow[i] = -INFINITY; lrow[i] = 0.f; }

  const float scale = 0.08838834764831845f;  // 1/sqrt(128)

  for (int k0 = 0; k0 < m0 + 16; k0 += 64) {   // causal bound: skip masked tiles
    // ---- scores S = Q K^T for 16 q-rows x 64 keys (four 16-col tiles) ----
    v8f sfr[4];
#pragma unroll
    for (int hf = 0; hf < 4; ++hf) sfr[hf] = v8f_zero();
#pragma unroll
    for (int hf = 0; hf < 4; ++hf) {
#pragma unroll
      for (int kc = 0; kc < 4; ++kc) {
        v16h bf = load_bT_f16(kb + (size_t)(k0 + hf * 16) * ldk + kc * 32, ldk, lane);
        sfr[hf] = __builtin_amdgcn_wmma_f32_16x16x32_f16(false, qf[kc], false, bf,
                                                         (short)0, sfr[hf], false, false);
      }
    }
    // ---- online softmax (row stats reduced across 16 lanes of a half-wave) ----
    float pvv[4][8], sc[8];
#pragma unroll
    for (int i = 0; i < 8; ++i) {
      const int row = m0 + rhalf + i;
      float t = -INFINITY;
#pragma unroll
      for (int hf = 0; hf < 4; ++hf) {
        float a0 = sfr[hf][i] * scale;
        if (k0 + hf * 16 + nidx > row) a0 = -INFINITY;   // causal mask
        pvv[hf][i] = a0;
        t = fmaxf(t, a0);
      }
      t = fmaxf(t, __shfl_xor(t, 1, 32));
      t = fmaxf(t, __shfl_xor(t, 2, 32));
      t = fmaxf(t, __shfl_xor(t, 4, 32));
      t = fmaxf(t, __shfl_xor(t, 8, 32));
      const float mn = fmaxf(mrow[i], t);                // finite: k0 <= m0 <= row
      float rs = 0.f;
#pragma unroll
      for (int hf = 0; hf < 4; ++hf) {
        const float p = __expf(pvv[hf][i] - mn);         // exp(-inf)=0 for masked
        pvv[hf][i] = p;
        rs += p;
      }
      rs += __shfl_xor(rs, 1, 32);
      rs += __shfl_xor(rs, 2, 32);
      rs += __shfl_xor(rs, 4, 32);
      rs += __shfl_xor(rs, 8, 32);
      const float scl = __expf(mrow[i] - mn);
      lrow[i] = lrow[i] * scl + rs;
      mrow[i] = mn;
      sc[i]  = scl;
    }
#pragma unroll
    for (int dt = 0; dt < 8; ++dt)
#pragma unroll
      for (int i = 0; i < 8; ++i) acc[dt][i] *= sc[i];

    // ---- relayout P (C-frag layout) -> two A-fragments via LDS ----
#pragma unroll
    for (int hf = 0; hf < 4; ++hf)
#pragma unroll
      for (int i = 0; i < 8; ++i)
        pl[(rhalf + i) * 64 + hf * 16 + nidx] = (_Float16)pvv[hf][i];
    __syncthreads();
    v16h pa0, pa1;
    {
      const _Float16* pr = pl + (size_t)(lane & 15) * 64 + ((lane >> 4) << 3);
#pragma unroll
      for (int i = 0; i < 8; ++i) {
        pa0[i]     = pr[i];
        pa0[8 + i] = pr[16 + i];
        pa1[i]     = pr[32 + i];
        pa1[8 + i] = pr[48 + i];
      }
    }
    // ---- O += P V: B-frags from V^T are per-lane contiguous along keys ----
#pragma unroll
    for (int dt = 0; dt < 8; ++dt) {
      v16h bv0 = load_bT_f16(vtb + (size_t)(dt * 16) * MTOT_ + k0,      MTOT_, lane);
      v16h bv1 = load_bT_f16(vtb + (size_t)(dt * 16) * MTOT_ + k0 + 32, MTOT_, lane);
      acc[dt] = __builtin_amdgcn_wmma_f32_16x16x32_f16(false, pa0, false, bv0,
                                                       (short)0, acc[dt], false, false);
      acc[dt] = __builtin_amdgcn_wmma_f32_16x16x32_f16(false, pa1, false, bv1,
                                                       (short)0, acc[dt], false, false);
    }
    __syncthreads();
  }

  // ---- normalize and store O (f16) ----
  float rinv[8];
#pragma unroll
  for (int i = 0; i < 8; ++i) rinv[i] = 1.f / lrow[i];
  _Float16* ob = O + ((size_t)b * S_ + m0) * ldq + h * D_;
#pragma unroll
  for (int dt = 0; dt < 8; ++dt) {
#pragma unroll
    for (int i = 0; i < 8; ++i) {
      const float o = acc[dt][i] * rinv[i];
      ob[(size_t)(rhalf + i) * ldq + dt * 16 + nidx] = (_Float16)o;
    }
  }
}

// ---------------------------------------------------------------------------
// Host launcher
// ---------------------------------------------------------------------------
extern "C" void kernel_launch(void* const* d_in, const int* in_sizes, int n_in,
                              void* d_out, int out_size, void* d_ws, size_t ws_size,
                              hipStream_t stream) {
  (void)in_sizes; (void)n_in; (void)out_size; (void)ws_size;
  const float* x  = (const float*)d_in[0];   // (B,S,HID) f32
  const float* fc = (const float*)d_in[1];   // (S,D/2,2) f32
  const float* wq = (const float*)d_in[2];   // (HID, H*D)
  const float* wk = (const float*)d_in[3];   // (HID, KVH*D)
  const float* wv = (const float*)d_in[4];   // (HID, KVH*D)
  const float* wo = (const float*)d_in[5];   // (H*D, HID)
  float* out = (float*)d_out;

  const int M = MTOT_;                       // 4096 token rows
  const int NQ = H_ * D_;                    // 2048
  const int NKV = KVH_ * D_;                 // 512

  // Workspace layout (all f16)
  char* ws = (char*)d_ws;
  size_t off = 0;
  _Float16* xh  = (_Float16*)(ws + off); off += (size_t)M * HID_ * 2;     // 16 MiB
  _Float16* wqT = (_Float16*)(ws + off); off += (size_t)NQ * HID_ * 2;    //  8 MiB
  _Float16* wkT = (_Float16*)(ws + off); off += (size_t)NKV * HID_ * 2;   //  2 MiB
  _Float16* wvT = (_Float16*)(ws + off); off += (size_t)NKV * HID_ * 2;   //  2 MiB
  _Float16* woT = (_Float16*)(ws + off); off += (size_t)HID_ * NQ * 2;    //  8 MiB
  _Float16* qf  = (_Float16*)(ws + off); off += (size_t)M * NQ * 2;       // 16 MiB
  _Float16* kf  = (_Float16*)(ws + off); off += (size_t)M * NKV * 2;      //  4 MiB
  _Float16* vT  = (_Float16*)(ws + off); off += (size_t)NKV * M * 2;      //  4 MiB
  _Float16* of  = (_Float16*)(ws + off); off += (size_t)M * NQ * 2;       // 16 MiB

  // 0) Pre-pass: x -> f16; weights -> f16 transposed (N-major)
  {
    const size_t nx = (size_t)M * HID_;
    cvt_f32_to_f16_kernel<<<(unsigned)(nx / 4 / 256), 256, 0, stream>>>(x, xh, nx);
    dim3 blk(32, 8);
    transpose_cvt_kernel<<<dim3(NQ / 32,  HID_ / 32), blk, 0, stream>>>(wq, wqT, HID_, NQ);
    transpose_cvt_kernel<<<dim3(NKV / 32, HID_ / 32), blk, 0, stream>>>(wk, wkT, HID_, NKV);
    transpose_cvt_kernel<<<dim3(NKV / 32, HID_ / 32), blk, 0, stream>>>(wv, wvT, HID_, NKV);
    transpose_cvt_kernel<<<dim3(HID_ / 32, NQ / 32),  blk, 0, stream>>>(wo, woT, NQ, HID_);
  }

  // 1) QKV projections (V stored transposed for attention PV stage)
  gemm_f16T_f16<false><<<dim3(NQ / 64,  M / 64), 32, 0, stream>>>(xh, wqT, qf, M, NQ,  HID_);
  gemm_f16T_f16<false><<<dim3(NKV / 64, M / 64), 32, 0, stream>>>(xh, wkT, kf, M, NKV, HID_);
  gemm_f16T_f16<true ><<<dim3(NKV / 64, M / 64), 32, 0, stream>>>(xh, wvT, vT, M, NKV, HID_);

  // 2) RoPE on Q and K
  {
    const size_t npq = (size_t)M * H_ * (D_ / 2);
    rope_f16_kernel<<<(unsigned)((npq + 255) / 256), 256, 0, stream>>>(qf, fc, H_);
    const size_t npk = (size_t)M * KVH_ * (D_ / 2);
    rope_f16_kernel<<<(unsigned)((npk + 255) / 256), 256, 0, stream>>>(kf, fc, KVH_);
  }

  // 3) Causal GQA flash attention
  flash_attn_kernel<<<dim3(S_ / 16, H_, B_), 32, 0, stream>>>(qf, kf, vT, of);

  // 4) Output projection (f16 x f16^T -> f32)
  gemm_f16T_f32<<<dim3(HID_ / 64, M / 64), 32, 0, stream>>>(of, woT, out, M, HID_, NQ);
}
